// VariableSelectionNetwork_28793460753141
// MI455X (gfx1250) — matrix operand order repaired and checked by reading, not compiled, not run on hardware
//
#include <hip/hip_runtime.h>
#include <math.h>

// Problem dims (from reference): B=32, T=512 -> M=16384 tokens, NV=16, D=DH=256
#define M_TOK   16384
#define NVAR    16
#define DDIM    256
#define DHID    256
#define TM      64          // tokens per workgroup in fused kernel
#define LDA     264         // padded LDS row stride (halves), 528B = 16B-aligned

typedef __attribute__((ext_vector_type(16))) __bf16 v16bf;
typedef __attribute__((ext_vector_type(8)))  __bf16 v8bf;
typedef __attribute__((ext_vector_type(8)))  float  v8f;

static __device__ __forceinline__ __bf16 f2bf(float f) {
  unsigned u = __float_as_uint(f);
  unsigned r = u + 0x7FFFu + ((u >> 16) & 1u);   // round-to-nearest-even
  unsigned short h = (unsigned short)(r >> 16);
  return __builtin_bit_cast(__bf16, h);
}
static __device__ __forceinline__ float bf2f(__bf16 b) {
  unsigned short h = __builtin_bit_cast(unsigned short, b);
  return __uint_as_float(((unsigned)h) << 16);
}
static __device__ __forceinline__ v8f vzero8() {
  v8f r;
#pragma unroll
  for (int i = 0; i < 8; ++i) r[i] = 0.f;
  return r;
}
static __device__ __forceinline__ v8f wmma_bf16(v16bf a, v16bf b, v8f c) {
  // D(16x16,f32) = A(16x32,bf16) * B(32x16,bf16) + C
  return __builtin_amdgcn_wmma_f32_16x16x32_bf16(false, a, false, b, (short)0, c,
                                                 false, false);
}

// ---------------------------------------------------------------------------
// Weight pre-pass: fp32 [f][k][n] -> bf16 [f][n][k] (so B fragments are two 8-half
// runs contiguous along K in LDS).
// ---------------------------------------------------------------------------
__global__ __launch_bounds__(256) void vsn_wcvt(const float* __restrict__ in,
                                               __bf16* __restrict__ out,
                                               int F, int K, int N) {
  long total = (long)F * K * N;
  long kn = (long)K * N;
  for (long i = (long)blockIdx.x * blockDim.x + threadIdx.x; i < total;
       i += (long)gridDim.x * blockDim.x) {
    long fi = i / kn;
    long rem = i - fi * kn;
    long n = rem / K;
    long k = rem - n * K;
    out[i] = f2bf(in[fi * kn + k * N + n]);
  }
}

// ---------------------------------------------------------------------------
// Weight GRN + LayerNorm + softmax over the NV raw variables.
// One token per thread; small weights cached in LDS. Writes w to d_out.
// ---------------------------------------------------------------------------
__global__ __launch_bounds__(256) void vsn_weight_grn(
    const float* __restrict__ x, const float* __restrict__ w1,
    const float* __restrict__ b1, const float* __restrict__ w2,
    const float* __restrict__ b2, const float* __restrict__ gw,
    const float* __restrict__ gb, const float* __restrict__ lg,
    const float* __restrict__ lb, float* __restrict__ wout) {
  __shared__ float s1[NVAR * DHID];
  __shared__ float sb1[DHID];
  __shared__ float s2[DHID * NVAR];
  __shared__ float sb2[NVAR];
  __shared__ float sgw[NVAR * 2 * NVAR];
  __shared__ float sgb[2 * NVAR];
  __shared__ float slg[NVAR];
  __shared__ float slb[NVAR];
  const int tid = threadIdx.x;
  for (int i = tid; i < NVAR * DHID; i += 256) s1[i] = w1[i];
  for (int i = tid; i < DHID; i += 256) sb1[i] = b1[i];
  for (int i = tid; i < DHID * NVAR; i += 256) s2[i] = w2[i];
  for (int i = tid; i < NVAR * 2 * NVAR; i += 256) sgw[i] = gw[i];
  if (tid < NVAR) { sb2[tid] = b2[tid]; slg[tid] = lg[tid]; slb[tid] = lb[tid]; }
  if (tid < 2 * NVAR) sgb[tid] = gb[tid];
  __syncthreads();

  const int t = blockIdx.x * 256 + tid;
  float xr[NVAR], y2[NVAR];
#pragma unroll
  for (int f = 0; f < NVAR; ++f) xr[f] = x[(size_t)t * NVAR + f];
#pragma unroll
  for (int f = 0; f < NVAR; ++f) y2[f] = sb2[f];
  for (int h = 0; h < DHID; ++h) {        // fused fc1 -> elu -> fc2
    float s = sb1[h];
#pragma unroll
    for (int f = 0; f < NVAR; ++f) s = fmaf(xr[f], s1[f * DHID + h], s);
    s = s > 0.f ? s : (__expf(s) - 1.f);
#pragma unroll
    for (int f = 0; f < NVAR; ++f) y2[f] = fmaf(s, s2[h * NVAR + f], y2[f]);
  }
  float wp[2 * NVAR];
#pragma unroll
  for (int j = 0; j < 2 * NVAR; ++j) wp[j] = sgb[j];
#pragma unroll
  for (int f = 0; f < NVAR; ++f)
#pragma unroll
    for (int j = 0; j < 2 * NVAR; ++j)
      wp[j] = fmaf(y2[f], sgw[f * 2 * NVAR + j], wp[j]);

  float val[NVAR], mean = 0.f;
#pragma unroll
  for (int f = 0; f < NVAR; ++f) {
    float sig = 1.f / (1.f + __expf(-wp[NVAR + f]));
    val[f] = xr[f] + wp[f] * sig;
    mean += val[f];
  }
  mean *= (1.f / NVAR);
  float var = 0.f;
#pragma unroll
  for (int f = 0; f < NVAR; ++f) { float d = val[f] - mean; var += d * d; }
  var *= (1.f / NVAR);
  const float rstd = rsqrtf(var + 1e-5f);
  float mx = -1e30f;
#pragma unroll
  for (int f = 0; f < NVAR; ++f) {
    val[f] = (val[f] - mean) * rstd * slg[f] + slb[f];
    mx = fmaxf(mx, val[f]);
  }
  float den = 0.f;
#pragma unroll
  for (int f = 0; f < NVAR; ++f) { val[f] = __expf(val[f] - mx); den += val[f]; }
  const float inv = 1.f / den;
#pragma unroll
  for (int f = 0; f < NVAR; ++f) wout[(size_t)t * NVAR + f] = val[f] * inv;
}

// ---------------------------------------------------------------------------
// One K=256 WMMA GEMM over an LDS activation tile [TM][LDA] (bf16) against a
// global bf16 weight matrix stored [256 n][256 k]. Wave layout: 8 waves,
// wave = (mtile<<1)|nhalf, each wave owns a [16,128] output strip (8 v8f accs).
// Weight k-slices are VGPR-prefetched and staged through LDS (sWt [256][32]).
// ---------------------------------------------------------------------------
static __device__ __forceinline__ void gemm_k256(
    const __bf16* __restrict__ act, const __bf16* __restrict__ wgl,
    __bf16* __restrict__ wt, int row, int khalf, int nhalf, int l15, int tid,
    v8f acc[8]) {
#pragma unroll
  for (int nt = 0; nt < 8; ++nt) acc[nt] = vzero8();

  const uint4* g0 = (const uint4*)(wgl + (size_t)tid * 256);
  uint4 p0 = g0[0], p1 = g0[1], p2 = g0[2], p3 = g0[3];   // slice 0 prefetch

  for (int ks = 0; ks < 8; ++ks) {
    __syncthreads();                       // previous readers of wt done
    uint4* wrow = (uint4*)(wt + tid * 32); // row n = tid, 32 halves
    wrow[0] = p0; wrow[1] = p1; wrow[2] = p2; wrow[3] = p3;
    __syncthreads();
    if (ks < 7) {                          // prefetch next k-slice
      const uint4* gn = (const uint4*)(wgl + (size_t)tid * 256 + (ks + 1) * 32);
      p0 = gn[0]; p1 = gn[1]; p2 = gn[2]; p3 = gn[3];
    }
    const int kk = ks * 32;
    // A fragment: row = mtile*16 + l15; K runs [khalf*8,+8) and [16+khalf*8,+8)
    v8bf alo = *(const v8bf*)(act + row * LDA + kk + khalf * 8);
    v8bf ahi = *(const v8bf*)(act + row * LDA + kk + 16 + khalf * 8);
    v16bf a = __builtin_shufflevector(alo, ahi, 0, 1, 2, 3, 4, 5, 6, 7, 8, 9,
                                      10, 11, 12, 13, 14, 15);
#pragma unroll
    for (int nt = 0; nt < 8; ++nt) {
      const int col = nhalf * 128 + nt * 16 + l15;
      // B fragment: col fixed, K runs [khalf*8,+8) and [16+khalf*8,+8) (transposed W)
      v8bf blo = *(const v8bf*)(wt + col * 32 + khalf * 8);
      v8bf bhi = *(const v8bf*)(wt + col * 32 + 16 + khalf * 8);
      v16bf b = __builtin_shufflevector(blo, bhi, 0, 1, 2, 3, 4, 5, 6, 7, 8, 9,
                                        10, 11, 12, 13, 14, 15);
      acc[nt] = wmma_bf16(a, b, acc[nt]);
    }
  }
}

// ---------------------------------------------------------------------------
// Fused per-variable GRN chain + residual + LayerNorm + softmax-weighted sum.
// Grid: 256 workgroups x 256 threads; each WG owns 64 tokens, loops f=0..15,
// accumulates z in registers, stores z once.
// ---------------------------------------------------------------------------
__global__ __launch_bounds__(256) void vsn_fused(
    const float* __restrict__ x, const float* __restrict__ enc_w,
    const float* __restrict__ enc_b, const float* __restrict__ fc1b,
    const float* __restrict__ fc2b, const float* __restrict__ glub,
    const float* __restrict__ lng, const float* __restrict__ lnb,
    const __bf16* __restrict__ wsbf, const float* __restrict__ wsel,
    float* __restrict__ zout) {
  extern __shared__ __align__(32) unsigned char smem[];
  __bf16* sActA = (__bf16*)smem;            // [TM][LDA]
  __bf16* sActB = sActA + TM * LDA;         // [TM][LDA]
  __bf16* sWt   = sActB + TM * LDA;         // [256][32] weight k-slice (transposed)
  float* sSum = (float*)(sWt + 256 * 32);   // [TM] LN row sums
  float* sSq  = sSum + TM;                  // [TM]
  float* sX   = sSq + TM;                   // [TM] x[token][f]
  float* sW   = sX + TM;                    // [TM] softmax weight w[token][f]

  const int tid = threadIdx.x;
  const int lane = tid & 31, wave = tid >> 5;
  const int mtile = wave >> 1, nhalf = wave & 1;
  const int khalf = lane >> 4, l15 = lane & 15;
  const int row = mtile * 16 + l15;
  const int tile0 = blockIdx.x * TM;

  const __bf16* fc1T = wsbf;
  const __bf16* fc2T = wsbf + (size_t)NVAR * DDIM * DHID;
  const __bf16* gluT = wsbf + (size_t)2 * NVAR * DDIM * DHID;

  v8f zacc[8];
#pragma unroll
  for (int nt = 0; nt < 8; ++nt) zacc[nt] = vzero8();

  for (int f = 0; f < NVAR; ++f) {
    __syncthreads();
    if (tid < TM) {
      sX[tid] = x[(size_t)(tile0 + tid) * NVAR + f];
      sW[tid] = wsel[(size_t)(tile0 + tid) * NVAR + f];
    } else if (tid < 2 * TM) {
      sSum[tid - TM] = 0.f;
    } else if (tid < 3 * TM) {
      sSq[tid - 2 * TM] = 0.f;
    }
    __builtin_prefetch(fc1T + (size_t)f * DDIM * DHID + (size_t)tid * 64, 0, 1);
    __syncthreads();

    // encoder: h = relu(x*enc_w + enc_b) -> sActA (bf16)
    for (int e = tid; e < TM * DDIM; e += 256) {
      const int rw = e >> 8, c = e & 255;
      float h = fmaf(sX[rw], enc_w[f * DDIM + c], enc_b[f * DDIM + c]);
      sActA[rw * LDA + c] = f2bf(fmaxf(h, 0.f));
    }

    v8f acc[8];

    // y1 = elu(h @ fc1 + b1) -> sActB
    gemm_k256(sActA, fc1T + (size_t)f * DDIM * DHID, sWt, row, khalf, nhalf, l15, tid, acc);
#pragma unroll
    for (int nt = 0; nt < 8; ++nt) {
      const int col = nhalf * 128 + nt * 16 + l15;
      const float bb = fc1b[f * DHID + col];
#pragma unroll
      for (int r = 0; r < 8; ++r) {
        float v = acc[nt][r] + bb;
        v = v > 0.f ? v : (__expf(v) - 1.f);
        sActB[(mtile * 16 + khalf * 8 + r) * LDA + col] = f2bf(v);
      }
    }

    // y2 = y1 @ fc2 + b2 -> sActA (h is dead; recomputed later from x)
    gemm_k256(sActB, fc2T + (size_t)f * DHID * DDIM, sWt, row, khalf, nhalf, l15, tid, acc);
#pragma unroll
    for (int nt = 0; nt < 8; ++nt) {
      const int col = nhalf * 128 + nt * 16 + l15;
      const float bb = fc2b[f * DDIM + col];
#pragma unroll
      for (int r = 0; r < 8; ++r)
        sActA[(mtile * 16 + khalf * 8 + r) * LDA + col] = f2bf(acc[nt][r] + bb);
    }

    // gate half of GLU: sg = sigmoid(y2 @ gluW[:, D:2D] + b) -> sActB
    gemm_k256(sActA, gluT + (size_t)f * DDIM * 2 * DDIM + (size_t)DDIM * DDIM,
              sWt, row, khalf, nhalf, l15, tid, acc);
#pragma unroll
    for (int nt = 0; nt < 8; ++nt) {
      const int col = nhalf * 128 + nt * 16 + l15;
      const float bb = glub[f * 2 * DDIM + DDIM + col];
#pragma unroll
      for (int r = 0; r < 8; ++r) {
        float v = acc[nt][r] + bb;
        sActB[(mtile * 16 + khalf * 8 + r) * LDA + col] =
            f2bf(1.f / (1.f + __expf(-v)));
      }
    }

    // linear half of GLU + residual + LayerNorm + weighted accumulation
    gemm_k256(sActA, gluT + (size_t)f * DDIM * 2 * DDIM, sWt, row, khalf, nhalf, l15, tid, acc);
    float xr[8], wr[8];
#pragma unroll
    for (int r = 0; r < 8; ++r) {
      const int rw = mtile * 16 + khalf * 8 + r;
      xr[r] = sX[rw];
      wr[r] = sW[rw];
    }
#pragma unroll
    for (int nt = 0; nt < 8; ++nt) {
      const int col = nhalf * 128 + nt * 16 + l15;
      const float bb = glub[f * 2 * DDIM + col];
      const float ew = enc_w[f * DDIM + col], eb = enc_b[f * DDIM + col];
#pragma unroll
      for (int r = 0; r < 8; ++r) {
        const int rw = mtile * 16 + khalf * 8 + r;
        const float av = acc[nt][r] + bb;
        const float h = fmaxf(fmaf(xr[r], ew, eb), 0.f);   // recompute encoder
        const float sg = bf2f(sActB[rw * LDA + col]);
        acc[nt][r] = h + av * sg;                           // pre-LN, in place
      }
    }
    // LN row stats: intra-wave shfl reduction over 16 lanes, cross-wave via
    // LDS ds_add_f32 (exactly 2 commutative adds per cell -> deterministic)
    float ms[8], rs[8];
#pragma unroll
    for (int r = 0; r < 8; ++r) {
      float s = 0.f, q = 0.f;
#pragma unroll
      for (int nt = 0; nt < 8; ++nt) { float v = acc[nt][r]; s += v; q += v * v; }
#pragma unroll
      for (int m = 1; m < 16; m <<= 1) {
        s += __shfl_xor(s, m, 32);
        q += __shfl_xor(q, m, 32);
      }
      if (l15 == 0) {
        const int rw = mtile * 16 + khalf * 8 + r;
        atomicAdd(&sSum[rw], s);
        atomicAdd(&sSq[rw], q);
      }
    }
    __syncthreads();
#pragma unroll
    for (int r = 0; r < 8; ++r) {
      const int rw = mtile * 16 + khalf * 8 + r;
      const float mean = sSum[rw] * (1.f / DDIM);
      const float var = sSq[rw] * (1.f / DDIM) - mean * mean;
      ms[r] = mean;
      rs[r] = rsqrtf(var + 1e-5f);
    }
#pragma unroll
    for (int nt = 0; nt < 8; ++nt) {
      const int col = nhalf * 128 + nt * 16 + l15;
      const float g = lng[f * DDIM + col], b = lnb[f * DDIM + col];
#pragma unroll
      for (int r = 0; r < 8; ++r) {
        const float hv = (acc[nt][r] - ms[r]) * rs[r] * g + b;
        zacc[nt][r] += wr[r] * hv;
      }
    }
  }

  // single coalesced-ish store of z
#pragma unroll
  for (int nt = 0; nt < 8; ++nt) {
    const int col = nhalf * 128 + nt * 16 + l15;
#pragma unroll
    for (int r = 0; r < 8; ++r) {
      const int tok = tile0 + mtile * 16 + khalf * 8 + r;
      zout[(size_t)tok * DDIM + col] = zacc[nt][r];
    }
  }
}

// ---------------------------------------------------------------------------
extern "C" void kernel_launch(void* const* d_in, const int* in_sizes, int n_in,
                              void* d_out, int out_size, void* d_ws,
                              size_t ws_size, hipStream_t stream) {
  const float* x      = (const float*)d_in[0];
  const float* enc_w  = (const float*)d_in[1];
  const float* enc_b  = (const float*)d_in[2];
  const float* fc1_w  = (const float*)d_in[3];
  const float* fc1_b  = (const float*)d_in[4];
  const float* fc2_w  = (const float*)d_in[5];
  const float* fc2_b  = (const float*)d_in[6];
  const float* glu_w  = (const float*)d_in[7];
  const float* glu_b  = (const float*)d_in[8];
  const float* ln_g   = (const float*)d_in[9];
  const float* ln_b   = (const float*)d_in[10];
  const float* wg1w   = (const float*)d_in[11];
  const float* wg1b   = (const float*)d_in[12];
  const float* wg2w   = (const float*)d_in[13];
  const float* wg2b   = (const float*)d_in[14];
  const float* wggw   = (const float*)d_in[15];
  const float* wggb   = (const float*)d_in[16];
  const float* wglg   = (const float*)d_in[17];
  const float* wglb   = (const float*)d_in[18];

  __bf16* ws = (__bf16*)d_ws;                  // bf16 transposed weights, 8 MB
  __bf16* fc1T = ws;
  __bf16* fc2T = ws + (size_t)NVAR * DDIM * DHID;
  __bf16* gluT = ws + (size_t)2 * NVAR * DDIM * DHID;

  float* zout = (float*)d_out;                 // z: [M_TOK][DDIM]
  float* wout = zout + (size_t)M_TOK * DDIM;   // w: [M_TOK][NVAR]

  vsn_wcvt<<<1024, 256, 0, stream>>>(fc1_w, fc1T, NVAR, DDIM, DHID);
  vsn_wcvt<<<1024, 256, 0, stream>>>(fc2_w, fc2T, NVAR, DHID, DDIM);
  vsn_wcvt<<<2048, 256, 0, stream>>>(glu_w, gluT, NVAR, DDIM, 2 * DDIM);

  vsn_weight_grn<<<M_TOK / 256, 256, 0, stream>>>(x, wg1w, wg1b, wg2w, wg2b,
                                                  wggw, wggb, wglg, wglb, wout);

  constexpr size_t SMEM =
      (size_t)(2 * TM * LDA + 256 * 32) * sizeof(__bf16) +
      (size_t)4 * TM * sizeof(float);          // 84,992 bytes
  vsn_fused<<<M_TOK / TM, 256, SMEM, stream>>>(x, enc_w, enc_b, fc1_b, fc2_b,
                                               glu_b, ln_g, ln_b, ws, wout,
                                               zout);
}
